// Decoder_35983236006126
// MI455X (gfx1250) — matrix operand order repaired and checked by reading, not compiled, not run on hardware
//
#include <hip/hip_runtime.h>
#include <hip/hip_bf16.h>
#include <stdint.h>

#define BATCH 16
#define TE 200
#define TD 300
#define NMELS 80
#define ENCD 512
#define DECD 1024
#define ATTND 128
#define LOCC 32
#define LOCK 31
#define PRE 256
#define NBLK 64

typedef unsigned short u16;
typedef __bf16 bf16;
typedef bf16 v16bf __attribute__((ext_vector_type(16)));
typedef float v8f __attribute__((ext_vector_type(8)));
typedef unsigned v4u __attribute__((ext_vector_type(4)));
typedef int v8i __attribute__((ext_vector_type(8)));
typedef int v4i __attribute__((ext_vector_type(4)));
typedef unsigned uv4 __attribute__((ext_vector_type(4)));  // plain vector for 16B loads

union Frag16 { v16bf v; u16 u[16]; uv4 q[2]; };
union FragC  { v8f v; float f[8]; };

// force global (AS1) 16-byte loads so fragment loads lower to global_load_b128, not flat_load
typedef __attribute__((address_space(1))) const uv4 uv4_g;
__device__ __forceinline__ uv4 gload16(const u16* p) {
  return *(uv4_g*)(uintptr_t)p;
}

__device__ __forceinline__ u16 f2bf(float f) {
  unsigned u = __builtin_bit_cast(unsigned, f);
  unsigned r = u + 0x7fffu + ((u >> 16) & 1u);
  return (u16)(r >> 16);
}
__device__ __forceinline__ float bf2f(u16 h) {
  unsigned u = ((unsigned)h) << 16;
  return __builtin_bit_cast(float, u);
}
__device__ __forceinline__ float sigf(float x) { return 1.0f / (1.0f + __expf(-x)); }

__device__ __forceinline__ v8f wmma_bf16(const Frag16& a, const Frag16& b, v8f c) {
#if defined(__HIP_DEVICE_COMPILE__)
  return __builtin_amdgcn_wmma_f32_16x16x32_bf16(false, a.v, false, b.v, (short)0, c, false, false);
#else
  return c;
#endif
}

// B fragment: 32x16 bf16 from row-major (N,K) weights in GLOBAL memory
__device__ __forceinline__ void loadB16(Frag16& b, const u16* W, int ldb, int n0, int k0, int lane) {
  int n = n0 + (lane & 15);
  int ks = k0 + ((lane < 16) ? 0 : 8);
  const u16* p = W + (size_t)n * ldb + ks;
  b.q[0] = gload16(p);
  b.q[1] = gload16(p + 16);
}
// B fragment from an LDS-resident 16x1024 row-major slice (rows = the tile's 16 N columns)
__device__ __forceinline__ void loadB16_lds(Frag16& b, const u16* Wl, int k0, int lane) {
  int r = lane & 15;
  int ks = k0 + ((lane < 16) ? 0 : 8);
  const u16* p = Wl + (size_t)r * DECD + ks;
  b.q[0] = *(const uv4*)p;
  b.q[1] = *(const uv4*)(p + 16);
}
// A fragment: 16x32 bf16 from row-major (16,K) activations in GLOBAL memory
__device__ __forceinline__ void loadA16(Frag16& a, const u16* A, int lda, int k0, int lane) {
  int r = lane & 15;
  int kc = k0 + ((lane < 16) ? 0 : 8);
  const u16* p = A + (size_t)r * lda + kc;
  a.q[0] = gload16(p);
  a.q[1] = gload16(p + 16);
}

__device__ __forceinline__ void grid_barrier(unsigned* cnt, unsigned gen) {
  __syncthreads();
  if (threadIdx.x == 0) {
    __threadfence();
    __hip_atomic_fetch_add(cnt, 1u, __ATOMIC_ACQ_REL, __HIP_MEMORY_SCOPE_AGENT);
    unsigned target = gen * gridDim.x;
    while (__hip_atomic_load(cnt, __ATOMIC_ACQUIRE, __HIP_MEMORY_SCOPE_AGENT) < target) {
#if defined(__HIP_DEVICE_COMPILE__)
      __builtin_amdgcn_s_sleep(2);
#endif
    }
  }
  __syncthreads();
}

// ---- Tensor Data Mover staging: one 16x1024 bf16 tile, global -> LDS ----
#if defined(__HIP_DEVICE_COMPILE__) && __has_builtin(__builtin_amdgcn_tensor_load_to_lds) && \
    __has_builtin(__builtin_amdgcn_s_wait_tensorcnt)
#define HAVE_TDM 1
#else
#define HAVE_TDM 0
#endif

__device__ __forceinline__ void tdm_load_16x1024(unsigned lds_byte_addr, const u16* gsrc) {
#if HAVE_TDM
  unsigned long long ga = (unsigned long long)(uintptr_t)gsrc;
  v4u g0;
  g0[0] = 1u;                                        // count=1, user descriptor
  g0[1] = lds_byte_addr;                             // lds_addr [63:32]
  g0[2] = (unsigned)(ga & 0xffffffffu);              // global_addr [95:64]
  g0[3] = (unsigned)((ga >> 32) & 0x01ffffffu)       // global_addr [120:96]
          | 0x80000000u;                             // type=2 ("image") [127:126]
  v8i g1;
  g1[0] = 0x00010000;                                // wg_mask=0, data_size=1 (2B)
  g1[1] = (int)(1024u << 16);                        // tensor_dim0=1024 (low 16 in [31:16])
  g1[2] = (int)(16u << 16);                          // tensor_dim0 hi=0; tensor_dim1=16
  g1[3] = (int)(1024u << 16);                        // tensor_dim1 hi=0; tile_dim0=1024
  g1[4] = 16;                                        // tile_dim1=16, tile_dim2=0
  g1[5] = 1024;                                      // tensor_dim0_stride low32
  g1[6] = 0;                                         // stride hi16, dim1_stride low16
  g1[7] = 0;
  v4i z4; z4[0] = z4[1] = z4[2] = z4[3] = 0;
#if defined(__clang_major__) && (__clang_major__ >= 23)
  v8i z8; z8[0]=z8[1]=z8[2]=z8[3]=z8[4]=z8[5]=z8[6]=z8[7]=0;
  __builtin_amdgcn_tensor_load_to_lds(g0, g1, z4, z4, z8, 0);
#else
  __builtin_amdgcn_tensor_load_to_lds(g0, g1, z4, z4, 0);
#endif
#else
  (void)lds_byte_addr; (void)gsrc;
#endif
}

// ---------------- prep kernels ----------------
__global__ void k_zero(unsigned* p, size_t n) {
  for (size_t i = (size_t)blockIdx.x * blockDim.x + threadIdx.x; i < n; i += (size_t)gridDim.x * blockDim.x)
    p[i] = 0u;
}
__global__ void k_f2bf(const float* s, u16* d, size_t n) {
  for (size_t i = (size_t)blockIdx.x * blockDim.x + threadIdx.x; i < n; i += (size_t)gridDim.x * blockDim.x)
    d[i] = f2bf(s[i]);
}
__global__ void k_pad2d(const float* s, u16* d, int rows, int kin, int kpad) {
  size_t n = (size_t)rows * kpad;
  for (size_t i = (size_t)blockIdx.x * blockDim.x + threadIdx.x; i < n; i += (size_t)gridDim.x * blockDim.x) {
    int r = (int)(i / kpad), k = (int)(i % kpad);
    d[i] = (k < kin) ? f2bf(s[(size_t)r * kin + k]) : (u16)0;
  }
}
__global__ void k_postw(const float* cw, u16* d, int Cout, int Cin, int Kp) {
  size_t n = (size_t)Cout * Kp;
  for (size_t i = (size_t)blockIdx.x * blockDim.x + threadIdx.x; i < n; i += (size_t)gridDim.x * blockDim.x) {
    int no = (int)(i / Kp), kk = (int)(i % Kp);
    u16 v = 0;
    if (kk < 5 * Cin) {
      int tap = kk / Cin, ci = kk - tap * Cin;
      v = f2bf(cw[((size_t)no * Cin + ci) * 5 + tap]);
    }
    d[i] = v;
  }
}

__global__ __launch_bounds__(128) void k_prenet1(const u16* melsbf, const u16* W, u16* X1) {
  const int t = blockIdx.x;
  const int lane = threadIdx.x & 31;
  const int nt = blockIdx.y * 4 + (threadIdx.x >> 5);
  FragC acc;
#pragma unroll
  for (int i = 0; i < 8; ++i) acc.f[i] = 0.0f;
  uv4 z = {0u, 0u, 0u, 0u};
  for (int ki = 0; ki < 3; ++ki) {
    const int k0 = ki * 32;
    Frag16 fa, fb;
    int r = lane & 15;
    int kc = k0 + ((lane < 16) ? 0 : 8);
    fa.q[0] = (t > 0 && kc < NMELS) ? gload16(melsbf + ((size_t)r * TD + (t - 1)) * NMELS + kc) : z;
    int kc2 = kc + 16;
    fa.q[1] = (t > 0 && kc2 < NMELS) ? gload16(melsbf + ((size_t)r * TD + (t - 1)) * NMELS + kc2) : z;
    loadB16(fb, W, 96, nt * 16, k0, lane);
    acc.v = wmma_bf16(fa, fb, acc.v);
  }
#pragma unroll
  for (int r8 = 0; r8 < 8; ++r8) {
    int m = r8 + ((lane < 16) ? 0 : 8);
    int n = nt * 16 + (lane & 15);
    X1[((size_t)t * BATCH + m) * PRE + n] = f2bf(fmaxf(acc.f[r8], 0.0f));
  }
}

__global__ __launch_bounds__(128) void k_prenet2(const u16* X1, const u16* W, u16* Xp) {
  const int t = blockIdx.x;
  const int lane = threadIdx.x & 31;
  const int nt = blockIdx.y * 4 + (threadIdx.x >> 5);
  const u16* Arow = X1 + (size_t)t * BATCH * PRE;
  FragC acc;
#pragma unroll
  for (int i = 0; i < 8; ++i) acc.f[i] = 0.0f;
  for (int ki = 0; ki < 8; ++ki) {
    Frag16 fa, fb;
    loadA16(fa, Arow, PRE, ki * 32, lane);
    loadB16(fb, W, PRE, nt * 16, ki * 32, lane);
    acc.v = wmma_bf16(fa, fb, acc.v);
  }
#pragma unroll
  for (int r8 = 0; r8 < 8; ++r8) {
    int m = r8 + ((lane < 16) ? 0 : 8);
    int n = nt * 16 + (lane & 15);
    Xp[((size_t)t * BATCH + m) * PRE + n] = f2bf(fmaxf(acc.f[r8], 0.0f));
  }
}

__global__ __launch_bounds__(128) void k_hcache(const u16* encbf, const u16* W, const float* bias, float* hc) {
  const int mt = blockIdx.x;
  const int lane = threadIdx.x & 31;
  const int nt = blockIdx.y * 4 + (threadIdx.x >> 5);
  const u16* Arow = encbf + (size_t)mt * 16 * ENCD;
  FragC acc;
#pragma unroll
  for (int i = 0; i < 8; ++i) acc.f[i] = 0.0f;
  for (int ki = 0; ki < 16; ++ki) {
    Frag16 fa, fb;
    loadA16(fa, Arow, ENCD, ki * 32, lane);
    loadB16(fb, W, ENCD, nt * 16, ki * 32, lane);
    acc.v = wmma_bf16(fa, fb, acc.v);
  }
#pragma unroll
  for (int r8 = 0; r8 < 8; ++r8) {
    int m = r8 + ((lane < 16) ? 0 : 8);
    int n = nt * 16 + (lane & 15);
    hc[((size_t)mt * 16 + m) * ATTND + n] = acc.f[r8] + bias[n];
  }
}

// ---------------- persistent decoder scan ----------------
struct ScanArgs {
  const float* enc; const float* hcache;
  const unsigned char* emask; const unsigned char* mmask;
  const u16 *Xpre, *W0ih, *W0hh, *W1ih, *W1hh, *Wmel;
  const float *bih0, *bhh0, *bih1, *bhh1;
  const float *dpw, *lcw, *lcb, *lpw, *fpw, *melb, *stopw, *stopb;
  u16 *h0bf, *h1bf, *ctxbf, *act0;
  float *aw, *caw;
  float *out_mel, *out_stop, *out_attn;
  unsigned* bar;
};

__global__ __launch_bounds__(256) void k_scan(ScanArgs A) {
  // dynamic LDS: weight-stationary slices [2 matrices][4 gates][16 rows][1024 k] bf16 = 256 KB
  extern __shared__ u16 ldsw[];
  __shared__ float gpart[8][16][16];
  __shared__ float c0s[16][16];
  __shared__ float c1s[16][16];
  __shared__ float s_att[ATTND];
  __shared__ float esh[256];
  __shared__ float red[256];
  __shared__ float wsh[256];

  const int tid = threadIdx.x;
  const int lane = tid & 31;
  const int wv = tid >> 5;
  const int blk = blockIdx.x;
  const int j0 = blk * 16;

  c0s[tid >> 4][tid & 15] = 0.0f;
  c1s[tid >> 4][tid & 15] = 0.0f;

  // ---- stage this WG's W0hh / W1hh slices into LDS (TDM if available) ----
#if HAVE_TDM
  if (wv == 0) {
#pragma unroll
    for (int m = 0; m < 2; ++m) {
      const u16* Wsrc = (m == 0) ? A.W0hh : A.W1hh;
#pragma unroll
      for (int g = 0; g < 4; ++g) {
        unsigned dst = (unsigned)(uintptr_t)&ldsw[((size_t)(m * 4 + g) * 16) * DECD];
        tdm_load_16x1024(dst, Wsrc + (size_t)(g * DECD + j0) * DECD);
      }
    }
    __builtin_amdgcn_s_wait_tensorcnt(0);
  }
#else
  for (int m = 0; m < 2; ++m) {
    const u16* Wsrc = (m == 0) ? A.W0hh : A.W1hh;
    for (int g = 0; g < 4; ++g)
      for (int i = tid; i < 16 * DECD; i += 256)
        ldsw[((size_t)(m * 4 + g) * 16) * DECD + i] =
            Wsrc[(size_t)(g * DECD + j0 + (i >> 10)) * DECD + (i & (DECD - 1))];
  }
#endif
  __syncthreads();

  unsigned gen = 0;
  for (int t = 0; t <= TD; ++t) {
    const int p = t & 1, q = p ^ 1;
    const u16* h0q = A.h0bf + q * BATCH * DECD;
    u16*       h0p = A.h0bf + p * BATCH * DECD;
    const u16* h1q = A.h1bf + q * BATCH * DECD;
    u16*       h1p = A.h1bf + p * BATCH * DECD;
    const u16* ctxq = A.ctxbf + q * BATCH * ENCD;
    u16*       ctxp = A.ctxbf + p * BATCH * ENCD;
    const float* awq = A.aw + q * BATCH * TE;
    float*       awp = A.aw + p * BATCH * TE;
    const float* cawq = A.caw + q * BATCH * TE;
    float*       cawp = A.caw + p * BATCH * TE;

    // ---- PHASE A: attention(t) blocks 0..15; mel(t-1) block 16; stop(t-1) block 17 ----
    if (t < TD && blk < BATCH) {
      const int b = blk;
      if (tid < ATTND) {
        float acc = 0.0f;
        const u16* h1r = h1q + b * DECD;
        const float* dw = A.dpw + (size_t)tid * DECD;
        for (int k = 0; k < DECD; ++k) acc += bf2f(h1r[k]) * dw[k];
        s_att[tid] = acc;
      }
      __syncthreads();
      float e_val = -__builtin_inff();
      if (tid < TE) {
        float fc[LOCC];
#pragma unroll
        for (int c = 0; c < LOCC; ++c) fc[c] = A.lcb[c];
        for (int k = 0; k < LOCK; ++k) {
          int tt = tid + k - (LOCK / 2);
          if (tt >= 0 && tt < TE) {
            float av = awq[b * TE + tt];
            float cv = cawq[b * TE + tt];
#pragma unroll
            for (int c = 0; c < LOCC; ++c)
              fc[c] += av * A.lcw[(c * 2) * LOCK + k] + cv * A.lcw[(c * 2 + 1) * LOCK + k];
          }
        }
        float acc = 0.0f;
        const float* hcr = A.hcache + ((size_t)b * TE + tid) * ATTND;
        for (int a = 0; a < ATTND; ++a) {
          float fl = 0.0f;
          const float* lp = A.lpw + a * LOCC;
#pragma unroll
          for (int c = 0; c < LOCC; ++c) fl += fc[c] * lp[c];
          acc += tanhf(hcr[a] + s_att[a] + fl) * A.fpw[a];
        }
        e_val = A.emask[b * TE + tid] ? acc : -__builtin_inff();
      }
      esh[tid] = e_val;
      red[tid] = e_val;
      __syncthreads();
      for (int s = 128; s > 0; s >>= 1) {
        if (tid < s) red[tid] = fmaxf(red[tid], red[tid + s]);
        __syncthreads();
      }
      float mx = red[0];
      __syncthreads();
      float ex = (tid < TE) ? __expf(esh[tid] - mx) : 0.0f;
      red[tid] = ex;
      __syncthreads();
      for (int s = 128; s > 0; s >>= 1) {
        if (tid < s) red[tid] += red[tid + s];
        __syncthreads();
      }
      float wgt = ex / red[0];
      wsh[tid] = (tid < TE) ? wgt : 0.0f;
      if (tid < TE) {
        A.out_attn[((size_t)b * TD + t) * TE + tid] = wgt;
        awp[b * TE + tid] = wgt;
        cawp[b * TE + tid] = cawq[b * TE + tid] + wgt;
      }
      __syncthreads();
      for (int j = tid; j < ENCD; j += 256) {
        float acc = 0.0f;
        const float* er = A.enc + ((size_t)b * TE) * ENCD + j;
        for (int tt = 0; tt < TE; ++tt) acc += wsh[tt] * er[(size_t)tt * ENCD];
        ctxp[b * ENCD + j] = f2bf(acc);
      }
    } else if (t > 0 && blk == BATCH) {
      const int tp = t - 1;
      if (wv < 5) {
        FragC acc;
#pragma unroll
        for (int i = 0; i < 8; ++i) acc.f[i] = 0.0f;
        const int nb = wv * 16;
        for (int ki = 0; ki < 48; ++ki) {
          const int k0 = ki * 32;
          Frag16 fa, fb;
          {
            int r = lane & 15;
            int kc = k0 + ((lane < 16) ? 0 : 8);
            const u16* p0 = (kc < DECD) ? (h1q + (size_t)r * DECD + kc) : (ctxq + (size_t)r * ENCD + (kc - DECD));
            int kc2 = kc + 16;
            const u16* p1 = (kc2 < DECD) ? (h1q + (size_t)r * DECD + kc2) : (ctxq + (size_t)r * ENCD + (kc2 - DECD));
            fa.q[0] = gload16(p0);
            fa.q[1] = gload16(p1);
          }
          loadB16(fb, A.Wmel, DECD + ENCD, nb, k0, lane);
          acc.v = wmma_bf16(fa, fb, acc.v);
        }
#pragma unroll
        for (int r8 = 0; r8 < 8; ++r8) {
          int m = r8 + ((lane < 16) ? 0 : 8);
          int n = nb + (lane & 15);
          float val = acc.f[r8] + A.melb[n];
          float mv = A.mmask[m * TD + tp] ? val : 0.0f;
          size_t o = ((size_t)m * TD + tp) * NMELS + n;
          A.out_mel[o] = mv;
          A.act0[o] = f2bf(mv);
        }
      }
    } else if (t > 0 && blk == BATCH + 1) {
      const int tp = t - 1;
      int b = tid >> 4, part = tid & 15;
      float acc = 0.0f;
      for (int k = part * 96; k < part * 96 + 96; ++k) {
        float x = (k < DECD) ? bf2f(h1q[b * DECD + k]) : bf2f(ctxq[b * ENCD + (k - DECD)]);
        acc += x * A.stopw[k];
      }
      red[tid] = acc;
      __syncthreads();
      if (part == 0) {
        float s = A.stopb[0];
#pragma unroll
        for (int i = 0; i < 16; ++i) s += red[b * 16 + i];
        A.out_stop[b * TD + tp] = s;
      }
    }
    grid_barrier(A.bar, ++gen);
    if (t == TD) break;

    // ---- PHASE B: LSTM0 gates for this block's 16-column slice ----
    {
      const int g = wv & 3, half = wv >> 2;
      const int nrow = g * DECD + j0;
      const u16* XpreT = A.Xpre + (size_t)t * BATCH * PRE;
      const u16* Wl = &ldsw[(size_t)g * 16 * DECD];  // W0hh slice (m=0)
      FragC acc;
#pragma unroll
      for (int i = 0; i < 8; ++i) acc.f[i] = 0.0f;
      const int kiB = half ? 28 : 0, kiE = half ? 56 : 28;
      for (int ki = kiB; ki < kiE; ++ki) {
        Frag16 fa, fb;
        if (ki < 24) {  // rin = [prenet_x | ctx], K=768 (from global/L2)
          const int k0 = ki * 32;
          int r = lane & 15;
          int kc = k0 + ((lane < 16) ? 0 : 8);
          const u16* p0 = (kc < PRE) ? (XpreT + (size_t)r * PRE + kc) : (ctxp + (size_t)r * ENCD + (kc - PRE));
          int kc2 = kc + 16;
          const u16* p1 = (kc2 < PRE) ? (XpreT + (size_t)r * PRE + kc2) : (ctxp + (size_t)r * ENCD + (kc2 - PRE));
          fa.q[0] = gload16(p0);
          fa.q[1] = gload16(p1);
          loadB16(fb, A.W0ih, PRE + ENCD, nrow, k0, lane);
        } else {        // h0 recurrence, K=1024 (weights from LDS)
          const int k0 = (ki - 24) * 32;
          loadA16(fa, h0q, DECD, k0, lane);
          loadB16_lds(fb, Wl, k0, lane);
        }
        acc.v = wmma_bf16(fa, fb, acc.v);
      }
#pragma unroll
      for (int r8 = 0; r8 < 8; ++r8)
        gpart[wv][r8 + ((lane < 16) ? 0 : 8)][lane & 15] = acc.f[r8];
    }
    __syncthreads();
    {
      const int b = tid >> 4, jl = tid & 15;
      const int j = j0 + jl;
      float gi = gpart[0][b][jl] + gpart[4][b][jl] + A.bih0[j] + A.bhh0[j];
      float gf = gpart[1][b][jl] + gpart[5][b][jl] + A.bih0[DECD + j] + A.bhh0[DECD + j];
      float gg = gpart[2][b][jl] + gpart[6][b][jl] + A.bih0[2 * DECD + j] + A.bhh0[2 * DECD + j];
      float go = gpart[3][b][jl] + gpart[7][b][jl] + A.bih0[3 * DECD + j] + A.bhh0[3 * DECD + j];
      float c = sigf(gf) * c0s[b][jl] + sigf(gi) * tanhf(gg);
      c0s[b][jl] = c;
      h0p[b * DECD + j] = f2bf(sigf(go) * tanhf(c));
    }
    grid_barrier(A.bar, ++gen);

    // ---- PHASE C: LSTM1 ----
    {
      const int g = wv & 3, half = wv >> 2;
      const int nrow = g * DECD + j0;
      const u16* Wl = &ldsw[(size_t)(4 + g) * 16 * DECD];  // W1hh slice (m=1)
      FragC acc;
#pragma unroll
      for (int i = 0; i < 8; ++i) acc.f[i] = 0.0f;
      const int kiB = half ? 32 : 0, kiE = half ? 64 : 32;
      for (int ki = kiB; ki < kiE; ++ki) {
        Frag16 fa, fb;
        if (ki < 32) {
          loadA16(fa, h0p, DECD, ki * 32, lane);
          loadB16(fb, A.W1ih, DECD, nrow, ki * 32, lane);
        } else {
          loadA16(fa, h1q, DECD, (ki - 32) * 32, lane);
          loadB16_lds(fb, Wl, (ki - 32) * 32, lane);
        }
        acc.v = wmma_bf16(fa, fb, acc.v);
      }
#pragma unroll
      for (int r8 = 0; r8 < 8; ++r8)
        gpart[wv][r8 + ((lane < 16) ? 0 : 8)][lane & 15] = acc.f[r8];
    }
    __syncthreads();
    {
      const int b = tid >> 4, jl = tid & 15;
      const int j = j0 + jl;
      float gi = gpart[0][b][jl] + gpart[4][b][jl] + A.bih1[j] + A.bhh1[j];
      float gf = gpart[1][b][jl] + gpart[5][b][jl] + A.bih1[DECD + j] + A.bhh1[DECD + j];
      float gg = gpart[2][b][jl] + gpart[6][b][jl] + A.bih1[2 * DECD + j] + A.bhh1[2 * DECD + j];
      float go = gpart[3][b][jl] + gpart[7][b][jl] + A.bih1[3 * DECD + j] + A.bhh1[3 * DECD + j];
      float c = sigf(gf) * c1s[b][jl] + sigf(gi) * tanhf(gg);
      c1s[b][jl] = c;
      h1p[b * DECD + j] = f2bf(sigf(go) * tanhf(c));
    }
    grid_barrier(A.bar, ++gen);
  }
}

// ---------------- postnet ----------------
__global__ __launch_bounds__(128) void k_postconv(const u16* act, int Cin, int Kp, const u16* W,
                                                  int Cout, const float* cb, float* z) {
  const int mt = blockIdx.x;
  const int lane = threadIdx.x & 31;
  const int nt = blockIdx.y * 4 + (threadIdx.x >> 5);
  if (nt * 16 >= Cout) return;
  const int r0 = mt * 16;
  const int rr = r0 + (lane & 15);
  const int bb = rr / TD, t0 = rr % TD;
  const int Ktot = 5 * Cin;
  FragC acc;
#pragma unroll
  for (int i = 0; i < 8; ++i) acc.f[i] = 0.0f;
  uv4 zz = {0u, 0u, 0u, 0u};
  const int niter = Kp / 32;
  for (int ki = 0; ki < niter; ++ki) {
    const int k0 = ki * 32;
    Frag16 fa, fb;
#pragma unroll
    for (int h = 0; h < 2; ++h) {
      int kc = k0 + ((lane < 16) ? 0 : 8) + h * 16;
      uv4 v = zz;
      if (kc < Ktot) {
        int tap = kc / Cin, ci = kc - tap * Cin;
        int ts = t0 + tap - 2;
        if (ts >= 0 && ts < TD)
          v = gload16(act + ((size_t)(bb * TD + ts)) * Cin + ci);
      }
      fa.q[h] = v;
    }
    loadB16(fb, W, Kp, nt * 16, k0, lane);
    acc.v = wmma_bf16(fa, fb, acc.v);
  }
#pragma unroll
  for (int r8 = 0; r8 < 8; ++r8) {
    int m = r8 + ((lane < 16) ? 0 : 8);
    int grow = r0 + m;
    int n = nt * 16 + (lane & 15);
    z[(size_t)grow * Cout + n] = acc.f[r8] + cb[n];
  }
}

// 8 channels per block so lanes hit consecutive addresses
__global__ __launch_bounds__(256) void k_bnstats(const float* z, int Cout, float* stat) {
  __shared__ float rs[256], rq[256];
  const int chl = threadIdx.x & 7;
  const int ch = blockIdx.x * 8 + chl;
  const int r0 = threadIdx.x >> 3;
  float s = 0.0f, sq = 0.0f;
  for (int r = r0; r < BATCH * TD; r += 32) {
    float v = z[(size_t)r * Cout + ch];
    s += v; sq += v * v;
  }
  rs[threadIdx.x] = s; rq[threadIdx.x] = sq;
  __syncthreads();
  for (int st = 128; st >= 8; st >>= 1) {
    if (threadIdx.x < st) { rs[threadIdx.x] += rs[threadIdx.x + st]; rq[threadIdx.x] += rq[threadIdx.x + st]; }
    __syncthreads();
  }
  if (threadIdx.x < 8) {
    float mean = rs[threadIdx.x] / (float)(BATCH * TD);
    float var = rq[threadIdx.x] / (float)(BATCH * TD) - mean * mean;
    stat[ch] = mean;
    stat[Cout + ch] = rsqrtf(var + 1e-5f);
  }
}

__global__ void k_bnnorm(const float* z, const float* stat, const float* gamma, const float* beta,
                         int Cout, int last, u16* actout, const float* melouts,
                         const unsigned char* mmask, float* outpost) {
  size_t n = (size_t)BATCH * TD * Cout;
  for (size_t i = (size_t)blockIdx.x * blockDim.x + threadIdx.x; i < n; i += (size_t)gridDim.x * blockDim.x) {
    int ch = (int)(i % Cout);
    float v = gamma[ch] * (z[i] - stat[ch]) * stat[Cout + ch] + beta[ch];
    if (!last) {
      actout[i] = f2bf(tanhf(v));
    } else {
      size_t r = i / Cout;
      outpost[i] = mmask[r] ? (melouts[i] + v) : 0.0f;
    }
  }
}

// ---------------- host ----------------
extern "C" void kernel_launch(void* const* d_in, const int* in_sizes, int n_in,
                              void* d_out, int out_size, void* d_ws, size_t ws_size,
                              hipStream_t stream) {
  (void)in_sizes; (void)n_in; (void)out_size; (void)ws_size;
  const float* enc = (const float*)d_in[0];
  const unsigned char* emask = (const unsigned char*)d_in[1];
  const float* mels = (const float*)d_in[2];
  const unsigned char* mmask = (const unsigned char*)d_in[3];
  const float* pw1 = (const float*)d_in[4];
  const float* pw2 = (const float*)d_in[5];
  const float* epw = (const float*)d_in[6];
  const float* epb = (const float*)d_in[7];
  const float* dpw = (const float*)d_in[8];
  const float* lcw = (const float*)d_in[9];
  const float* lcb = (const float*)d_in[10];
  const float* lpw = (const float*)d_in[11];
  const float* fpw = (const float*)d_in[12];
  const float* w0ih = (const float*)d_in[13];
  const float* w0hh = (const float*)d_in[14];
  const float* b0ih = (const float*)d_in[15];
  const float* b0hh = (const float*)d_in[16];
  const float* w1ih = (const float*)d_in[17];
  const float* w1hh = (const float*)d_in[18];
  const float* b1ih = (const float*)d_in[19];
  const float* b1hh = (const float*)d_in[20];
  const float* melw = (const float*)d_in[21];
  const float* melb = (const float*)d_in[22];
  const float* stopw = (const float*)d_in[23];
  const float* stopb = (const float*)d_in[24];

  char* ws = (char*)d_ws;
  size_t off = 0;
  auto alloc = [&](size_t bytes) -> char* {
    char* pp = ws + off;
    off = (off + bytes + 255) & ~(size_t)255;
    return pp;
  };
  u16* W0IH = (u16*)alloc((size_t)4096 * 768 * 2);
  u16* W0HH = (u16*)alloc((size_t)4096 * 1024 * 2);
  u16* W1IH = (u16*)alloc((size_t)4096 * 1024 * 2);
  u16* W1HH = (u16*)alloc((size_t)4096 * 1024 * 2);
  u16* WMEL = (u16*)alloc((size_t)80 * 1536 * 2);
  u16* WPRE1 = (u16*)alloc((size_t)256 * 96 * 2);
  u16* WPRE2 = (u16*)alloc((size_t)256 * 256 * 2);
  u16* WENC = (u16*)alloc((size_t)128 * 512 * 2);
  u16* WPOST[5];
  int pcin[5] = {80, 512, 512, 512, 512};
  int pcout[5] = {512, 512, 512, 512, 80};
  int pkp[5] = {416, 2560, 2560, 2560, 2560};
  for (int i = 0; i < 5; ++i) WPOST[i] = (u16*)alloc((size_t)pcout[i] * pkp[i] * 2);
  u16* ENCBF = (u16*)alloc((size_t)BATCH * TE * ENCD * 2);
  u16* MELSBF = (u16*)alloc((size_t)BATCH * TD * NMELS * 2);
  u16* X1 = (u16*)alloc((size_t)TD * BATCH * PRE * 2);
  u16* XPRE = (u16*)alloc((size_t)TD * BATCH * PRE * 2);
  float* HCACHE = (float*)alloc((size_t)BATCH * TE * ATTND * 4);
  char* STATE = alloc((size_t)65536 + 65536 + 32768 + 25600 + 25600 + 256);
  u16* H0BF = (u16*)(STATE);
  u16* H1BF = (u16*)(STATE + 65536);
  u16* CTXBF = (u16*)(STATE + 131072);
  float* AW = (float*)(STATE + 163840);
  float* CAW = (float*)(STATE + 189440);
  unsigned* BAR = (unsigned*)(STATE + 215040);
  u16* ACT0 = (u16*)alloc((size_t)BATCH * TD * NMELS * 2);
  u16* ACTA = (u16*)alloc((size_t)BATCH * TD * 512 * 2);
  u16* ACTB = (u16*)alloc((size_t)BATCH * TD * 512 * 2);
  float* ZBUF = (float*)alloc((size_t)BATCH * TD * 512 * 4);
  float* BNSTAT = (float*)alloc((size_t)512 * 2 * 4);

  float* out_mel = (float*)d_out;
  float* out_post = out_mel + (size_t)BATCH * TD * NMELS;
  float* out_stop = out_mel + (size_t)2 * BATCH * TD * NMELS;
  float* out_attn = out_stop + (size_t)BATCH * TD;

  // ---- prep ----
  k_zero<<<256, 256, 0, stream>>>((unsigned*)STATE, (size_t)(215040 + 256) / 4);
  k_f2bf<<<1024, 256, 0, stream>>>(w0ih, W0IH, (size_t)4096 * 768);
  k_f2bf<<<1024, 256, 0, stream>>>(w0hh, W0HH, (size_t)4096 * 1024);
  k_f2bf<<<1024, 256, 0, stream>>>(w1ih, W1IH, (size_t)4096 * 1024);
  k_f2bf<<<1024, 256, 0, stream>>>(w1hh, W1HH, (size_t)4096 * 1024);
  k_f2bf<<<256, 256, 0, stream>>>(melw, WMEL, (size_t)80 * 1536);
  k_pad2d<<<64, 256, 0, stream>>>(pw1, WPRE1, 256, 80, 96);
  k_f2bf<<<128, 256, 0, stream>>>(pw2, WPRE2, (size_t)256 * 256);
  k_f2bf<<<128, 256, 0, stream>>>(epw, WENC, (size_t)128 * 512);
  k_f2bf<<<1024, 256, 0, stream>>>(enc, ENCBF, (size_t)BATCH * TE * ENCD);
  k_f2bf<<<512, 256, 0, stream>>>(mels, MELSBF, (size_t)BATCH * TD * NMELS);
  for (int i = 0; i < 5; ++i)
    k_postw<<<1024, 256, 0, stream>>>((const float*)d_in[25 + i * 4], WPOST[i], pcout[i], pcin[i], pkp[i]);

  k_prenet1<<<dim3(TD, 4), 128, 0, stream>>>(MELSBF, WPRE1, X1);
  k_prenet2<<<dim3(TD, 4), 128, 0, stream>>>(X1, WPRE2, XPRE);
  k_hcache<<<dim3(TE, 2), 128, 0, stream>>>(ENCBF, WENC, epb, HCACHE);

  // ---- persistent scan (64 WGs, 256 KB dynamic LDS for weight-stationary slices) ----
  ScanArgs SA;
  SA.enc = enc; SA.hcache = HCACHE; SA.emask = emask; SA.mmask = mmask;
  SA.Xpre = XPRE; SA.W0ih = W0IH; SA.W0hh = W0HH; SA.W1ih = W1IH; SA.W1hh = W1HH; SA.Wmel = WMEL;
  SA.bih0 = b0ih; SA.bhh0 = b0hh; SA.bih1 = b1ih; SA.bhh1 = b1hh;
  SA.dpw = dpw; SA.lcw = lcw; SA.lcb = lcb; SA.lpw = lpw; SA.fpw = fpw;
  SA.melb = melb; SA.stopw = stopw; SA.stopb = stopb;
  SA.h0bf = H0BF; SA.h1bf = H1BF; SA.ctxbf = CTXBF; SA.act0 = ACT0;
  SA.aw = AW; SA.caw = CAW;
  SA.out_mel = out_mel; SA.out_stop = out_stop; SA.out_attn = out_attn;
  SA.bar = BAR;
  const size_t scan_lds = (size_t)2 * 4 * 16 * DECD * sizeof(u16);  // 256 KB
  (void)hipFuncSetAttribute(reinterpret_cast<const void*>(&k_scan),
                            hipFuncAttributeMaxDynamicSharedMemorySize, (int)scan_lds);
  k_scan<<<NBLK, 256, scan_lds, stream>>>(SA);

  // ---- postnet ----
  const u16* actin = ACT0;
  u16* bufs[2] = {ACTA, ACTB};
  for (int L = 0; L < 5; ++L) {
    const float* cb = (const float*)d_in[25 + L * 4 + 1];
    const float* gamma = (const float*)d_in[25 + L * 4 + 2];
    const float* beta = (const float*)d_in[25 + L * 4 + 3];
    dim3 gconv(TD, (pcout[L] == 512) ? 8 : 2);
    k_postconv<<<gconv, 128, 0, stream>>>(actin, pcin[L], pkp[L], WPOST[L], pcout[L], cb, ZBUF);
    k_bnstats<<<pcout[L] / 8, 256, 0, stream>>>(ZBUF, pcout[L], BNSTAT);
    u16* aout = bufs[L & 1];
    k_bnnorm<<<2048, 256, 0, stream>>>(ZBUF, BNSTAT, gamma, beta, pcout[L], (L == 4) ? 1 : 0,
                                       aout, out_mel, mmask, out_post);
    actin = aout;
  }
}
